// SimpleNCDE_78383153152569
// MI455X (gfx1250) — matrix operand order repaired and checked by reading, not compiled, not run on hardware
//
#include <hip/hip_runtime.h>
#include <hip/hip_bf16.h>

// ---------------------------------------------------------------------------
// Neural CDE (GRU + RK4 ODE) persistent kernel for gfx1250 (MI455X).
// B=1024, T=201, IN=128, HID=512, OUT=64, 4 RK4 substeps.
// One workgroup owns 16 batch rows (the WMMA M tile) and runs the entire
// 200-step sequential scan internally. All GEMMs use v_wmma_f32_16x16x32_f16.
// Round 6: fully-unrolled straight-line software pipeline inside each GEMM
// slice. No loop-carried PHIs -> buffer rotation is SSA renaming (zero mov
// chains); sched fences cap in-flight prefetch at one fragment group.
// ---------------------------------------------------------------------------

typedef _Float16 f16;
typedef __attribute__((ext_vector_type(16))) _Float16 v16h;
typedef __attribute__((ext_vector_type(8)))  _Float16 v8h;
typedef __attribute__((ext_vector_type(8)))  float    v8f;

#define BDIM  256      // 8 wave32s
#define MT    16       // batch rows per workgroup (WMMA M)
#define HIDN  512
#define IND   128
#define NSUB  4

#if __has_builtin(__builtin_amdgcn_sched_barrier)
#define SCHED_FENCE() __builtin_amdgcn_sched_barrier(0)
#else
#define SCHED_FENCE()
#endif

union V16 { v16h v; v8h h[2]; };

// A fragment (16x32 f16) from LDS row-major [m][k] buffer.
// Layout: lane m=l&15; halves 0..15 -> k = k0+(l>>4)*16+0..15 (matches ldB).
__device__ __forceinline__ v16h ldA(const f16* a, int stride, int k0, int lane) {
  const f16* p = a + (lane & 15) * stride + k0 + ((lane >> 4) << 4);
  V16 r; r.h[0] = *(const v8h*)p; r.h[1] = *(const v8h*)(p + 8);
  return r.v;
}

// B fragment (32x16 f16) for D = A @ W^T with W stored row-major [N][K]:
// B[k][n] = W[n][k]; lane n=l&15 reads 16 contiguous halves at k0+(l>>4)*16.
__device__ __forceinline__ v16h ldB(const f16* w, int K, int nbase, int k0, int lane) {
  const f16* p = w + (size_t)(nbase + (lane & 15)) * K + k0 + ((lane >> 4) << 4);
  V16 r; r.h[0] = *(const v8h*)p; r.h[1] = *(const v8h*)(p + 8);
  return r.v;
}

__device__ __forceinline__ v8f wmma16(v16h a, v16h b, v8f c) {
  return __builtin_amdgcn_wmma_f32_16x16x32_f16(false, a, false, b, (short)0, c,
                                                false, false);
}

__device__ __forceinline__ float sigmoidf_(float x) {
  return 1.0f / (1.0f + __expf(-x));
}

// C layout: lane l, acc element i -> row (l>>4)*8+i, col nbase+(l&15).
__device__ __forceinline__ void stC_f32(float* dst, int stride, int nbase, v8f c, int lane) {
  int n = nbase + (lane & 15);
  int mb = (lane >> 4) * 8;
#pragma unroll
  for (int i = 0; i < 8; ++i) dst[(mb + i) * stride + n] = c[i];
}

// 16 x 64 slice of D = A[16xK] @ W^T per wave: 4 N tiles, 4 accumulators.
// Fully unrolled straight-line pipeline: fragments for step k+1 are issued
// before the WMMA group of step k; "rotation" is SSA renaming (no copies).
// Fences keep at most one prefetch group in flight per WMMA group.
template <int KITER>
__device__ __forceinline__ void gemm_slice(const f16* __restrict__ Asrc, int astride,
                                           const f16* __restrict__ W, int K,
                                           int nbase, int lane, v8f acc[4]) {
  v16h a_cur = ldA(Asrc, astride, 0, lane);
  v16h b_cur[4];
#pragma unroll
  for (int ti = 0; ti < 4; ++ti) b_cur[ti] = ldB(W, K, nbase + ti * 16, 0, lane);
#pragma unroll
  for (int kk = 0; kk < KITER; ++kk) {
    v16h a_nxt;
    v16h b_nxt[4];
    if (kk + 1 < KITER) {                       // compile-time condition
      a_nxt = ldA(Asrc, astride, (kk + 1) * 32, lane);
#pragma unroll
      for (int ti = 0; ti < 4; ++ti)
        b_nxt[ti] = ldB(W, K, nbase + ti * 16, (kk + 1) * 32, lane);
    }
    SCHED_FENCE();
#pragma unroll
    for (int ti = 0; ti < 4; ++ti) acc[ti] = wmma16(a_cur, b_cur[ti], acc[ti]);
    SCHED_FENCE();
    a_cur = a_nxt;                              // SSA rename, no instructions
#pragma unroll
    for (int ti = 0; ti < 4; ++ti) b_cur[ti] = b_nxt[ti];
  }
}

__global__ void __launch_bounds__(BDIM, 1)
ncde_persistent(const float* __restrict__ tp,
                const float* __restrict__ h0,
                const float* __restrict__ xs,
                const f16* __restrict__ Wih,   // [1536][128] f16
                const f16* __restrict__ Whh,   // [1536][512] f16
                const f16* __restrict__ W1w,   // [512][512] f16
                const f16* __restrict__ W2w,   // [512][512] f16
                const float* __restrict__ b_ih,
                const float* __restrict__ b_hh,
                const float* __restrict__ b1,
                const float* __restrict__ b2,
                const float* __restrict__ Wo,  // [64][512] f32
                const float* __restrict__ bo,
                float* __restrict__ out,
                int T) {
  extern __shared__ char smem[];
  f16*   xF   = (f16*)smem;                  // 16x128 f16   (x_t)
  f16*   hF   = xF + MT * IND;               // 16x512 f16   (y / h image, GEMM A)
  f16*   aF   = hF + MT * HIDN;              // 16x512 f16   (relu hidden)
  float* hS   = (float*)(aF + MT * HIDN);    // 16x512 f32   (h)
  float* accS = hS + MT * HIDN;              // 16x512 f32   (RK4 acc / h_n)
  float* kS   = accS + MT * HIDN;            // 16x512 f32   (func out / i_n)
  float* rzS  = kS + MT * HIDN;              // 16x1024 f32  (r,z gate pre-act)
  float* brz  = rzS + MT * 1024;             // 1024 (b_ih+b_hh for r,z)
  float* binS = brz + 1024;                  // 512
  float* bhnS = binS + HIDN;                 // 512
  float* b1S  = bhnS + HIDN;                 // 512
  float* b2S  = b1S + HIDN;                  // 512

  const int tid  = threadIdx.x;
  const int lane = tid & 31;
  const int wv   = tid >> 5;
  const int row0 = blockIdx.x * MT;
  const v8f vz = {0.f,0.f,0.f,0.f,0.f,0.f,0.f,0.f};

  // Stage biases + initial state once.
  for (int i = tid; i < 1024; i += BDIM) brz[i] = b_ih[i] + b_hh[i];
  for (int i = tid; i < HIDN; i += BDIM) {
    binS[i] = b_ih[1024 + i]; bhnS[i] = b_hh[1024 + i];
    b1S[i] = b1[i]; b2S[i] = b2[i];
  }
  for (int i = tid; i < MT * HIDN; i += BDIM) {
    float v = h0[(size_t)(row0 + (i >> 9)) * HIDN + (i & 511)];
    hS[i] = v; hF[i] = (f16)v;
  }
  __syncthreads();

#pragma unroll 1
  for (int t = 0; t < T - 1; ++t) {
    // ---- stage x_t ------------------------------------------------------
    const float* xg = xs + (size_t)t * 1024 * IND + (size_t)row0 * IND;
    for (int i = tid; i < MT * IND; i += BDIM) xF[i] = (f16)xg[i];
    const float dt  = tp[t + 1] - tp[t];
    const float sub = dt * (1.0f / NSUB);
    __syncthreads();

    // ---- GRU: fused [x|h] @ [Wih|Whh]^T for r,z cols (N=1024, K=640) ----
    // Two passes of 64 cols per wave -> only 4 accumulator tiles live.
#pragma unroll 1
    for (int pass = 0; pass < 2; ++pass) {
      const int n0w = wv * 128 + pass * 64;
      v8f acc[4] = {vz, vz, vz, vz};
      gemm_slice<4>(xF, IND, Wih, IND, n0w, lane, acc);      // K = 0..127 (x)
      gemm_slice<16>(hF, HIDN, Whh, HIDN, n0w, lane, acc);   // K = 128..639 (h)
#pragma unroll
      for (int ti = 0; ti < 4; ++ti) stC_f32(rzS, 1024, n0w + ti * 16, acc[ti], lane);
    }
    // ---- GRU: i_n (K=128) then h_n (K=512), kept separate (r gates h_n) -
    {
      const int n0w = wv * 64;
      v8f acc[4] = {vz, vz, vz, vz};
      gemm_slice<4>(xF, IND, Wih, IND, 1024 + n0w, lane, acc);
#pragma unroll
      for (int ti = 0; ti < 4; ++ti) stC_f32(kS, HIDN, n0w + ti * 16, acc[ti], lane);
    }
    {
      const int n0w = wv * 64;
      v8f acc[4] = {vz, vz, vz, vz};
      gemm_slice<16>(hF, HIDN, Whh, HIDN, 1024 + n0w, lane, acc);
#pragma unroll
      for (int ti = 0; ti < 4; ++ti) stC_f32(accS, HIDN, n0w + ti * 16, acc[ti], lane);
    }
    __syncthreads();

    // ---- GRU elementwise gate update -----------------------------------
    for (int i = tid; i < MT * HIDN; i += BDIM) {
      int m = i >> 9, n = i & 511;
      float r  = sigmoidf_(rzS[m * 1024 + n] + brz[n]);
      float z  = sigmoidf_(rzS[m * 1024 + 512 + n] + brz[512 + n]);
      float nn = tanhf(kS[i] + binS[n] + r * (accS[i] + bhnS[n]));
      float h  = (1.0f - z) * nn + z * hS[i];
      hS[i] = h; hF[i] = (f16)h;     // hF doubles as y for first func eval
    }
    __syncthreads();

    // ---- ODE: 4 x RK4, func(y) = relu(y@W1^T+b1)@W2^T+b2 ---------------
#pragma unroll 1
    for (int s = 0; s < NSUB; ++s) {
#pragma unroll 1
      for (int j = 0; j < 4; ++j) {
        const int n0w = wv * 64;
        // hidden = relu(y @ W1^T + b1) -> aF (f16)
        {
          v8f acc[4] = {vz, vz, vz, vz};
          gemm_slice<16>(hF, HIDN, W1w, HIDN, n0w, lane, acc);
          const int mb = (lane >> 4) * 8;
#pragma unroll
          for (int ti = 0; ti < 4; ++ti) {
            int n = n0w + ti * 16 + (lane & 15);
            float bb = b1S[n];
#pragma unroll
            for (int i = 0; i < 8; ++i) {
              float v = acc[ti][i] + bb;
              v = v > 0.f ? v : 0.f;
              aF[(mb + i) * HIDN + n] = (f16)v;
            }
          }
        }
        __syncthreads();
        // k_j = hidden @ W2^T + b2 -> kS (f32)
        {
          v8f acc[4] = {vz, vz, vz, vz};
          gemm_slice<16>(aF, HIDN, W2w, HIDN, n0w, lane, acc);
          const int mb = (lane >> 4) * 8;
#pragma unroll
          for (int ti = 0; ti < 4; ++ti) {
            int n = n0w + ti * 16 + (lane & 15);
            float bb = b2S[n];
#pragma unroll
            for (int i = 0; i < 8; ++i)
              kS[(mb + i) * HIDN + n] = acc[ti][i] + bb;
          }
        }
        __syncthreads();
        // RK4 stage combine
        for (int i = tid; i < MT * HIDN; i += BDIM) {
          float kv = kS[i];
          float h  = hS[i];
          if (j == 0) { accS[i] = kv;             hF[i] = (f16)(h + 0.5f * sub * kv); }
          else if (j == 1) { accS[i] += 2.f * kv; hF[i] = (f16)(h + 0.5f * sub * kv); }
          else if (j == 2) { accS[i] += 2.f * kv; hF[i] = (f16)(h + sub * kv); }
          else {
            h = h + (sub * (1.0f / 6.0f)) * (accS[i] + kv);
            hS[i] = h; hF[i] = (f16)h;
          }
        }
        __syncthreads();
      }
    }
  }

  // ---- output: out = h @ Wo^T + bo (tiny, once) ------------------------
  for (int i = tid; i < MT * 64; i += BDIM) {
    int m = i >> 6, o = i & 63;
    const float* wrow = Wo + o * HIDN;
    const float* hrow = hS + m * HIDN;
    float sacc = bo[o];
    for (int k2 = 0; k2 < HIDN; ++k2) sacc += hrow[k2] * wrow[k2];
    out[(size_t)(row0 + m) * 64 + o] = sacc;
  }
}

// f32 -> f16 weight conversion (once; weights then live in L2).
__global__ void cvt_f32_f16(const float* __restrict__ src, f16* __restrict__ dst, int n) {
  for (int i = blockIdx.x * blockDim.x + threadIdx.x; i < n; i += gridDim.x * blockDim.x)
    dst[i] = (f16)src[i];
}

extern "C" void kernel_launch(void* const* d_in, const int* in_sizes, int n_in,
                              void* d_out, int out_size, void* d_ws, size_t ws_size,
                              hipStream_t stream) {
  const float* tp  = (const float*)d_in[0];
  const float* h0  = (const float*)d_in[1];
  const float* xs  = (const float*)d_in[2];
  const float* Wih = (const float*)d_in[3];
  const float* bih = (const float*)d_in[4];
  const float* Whh = (const float*)d_in[5];
  const float* bhh = (const float*)d_in[6];
  const float* W1  = (const float*)d_in[7];
  const float* b1  = (const float*)d_in[8];
  const float* W2  = (const float*)d_in[9];
  const float* b2  = (const float*)d_in[10];
  const float* Wo  = (const float*)d_in[11];
  const float* bo  = (const float*)d_in[12];
  float* out = (float*)d_out;
  const int T = in_sizes[0];  // 201

  const size_t nWih = (size_t)3 * HIDN * IND;   // 196608
  const size_t nWhh = (size_t)3 * HIDN * HIDN;  // 786432
  const size_t nW1  = (size_t)HIDN * HIDN;      // 262144
  f16* wIh = (f16*)d_ws;
  f16* wHh = wIh + nWih;
  f16* w1h = wHh + nWhh;
  f16* w2h = w1h + nW1;

  cvt_f32_f16<<<256, 256, 0, stream>>>(Wih, wIh, (int)nWih);
  cvt_f32_f16<<<256, 256, 0, stream>>>(Whh, wHh, (int)nWhh);
  cvt_f32_f16<<<256, 256, 0, stream>>>(W1,  w1h, (int)nW1);
  cvt_f32_f16<<<256, 256, 0, stream>>>(W2,  w2h, (int)nW1);

  // LDS: f16 (x 4KB, h 16KB, a 16KB) + f32 (h/acc/k 96KB, rz 64KB, biases 12KB)
  const size_t lds =
      (size_t)(MT * IND + 2 * MT * HIDN) * sizeof(f16) +
      (size_t)(3 * MT * HIDN + MT * 1024 + 1024 + 4 * HIDN) * sizeof(float);

  ncde_persistent<<<dim3(1024 / MT), dim3(BDIM), lds, stream>>>(
      tp, h0, xs, wIh, wHh, w1h, w2h, bih, bhh, b1, b2, Wo, bo, out, T);
}
